// ROIRCNN_29025388986925
// MI455X (gfx1250) — matrix operand order repaired and checked by reading, not compiled, not run on hardware
//
#include <hip/hip_runtime.h>
#include <hip/hip_bf16.h>
#include <float.h>

// ---------------------------------------------------------------------------
// Types for CDNA5 WMMA (wave32): v_wmma_f32_16x16x32_bf16
// ---------------------------------------------------------------------------
typedef __attribute__((ext_vector_type(16))) __bf16 v16bf;
typedef __attribute__((ext_vector_type(8)))  __bf16 v8bf;
typedef __attribute__((ext_vector_type(8)))  float  v8f;
typedef __attribute__((ext_vector_type(4)))  float  v4f;

#define ROI_SIZE 7
#define N_CLASS 21
#define CCH 256
#define HF 50
#define WF 50
#define N_ROIS 256
#define IN_FLAT (CCH * ROI_SIZE * ROI_SIZE)   // 12544
#define HID 2048

// ---------------------------------------------------------------------------
// Kernel 1: torchvision-style RoIPool -> pooled activations as bf16
// pooled layout: A0[roi][c*49 + ph*7 + pw]  (== pooled.reshape(N_ROIS, -1))
// ---------------------------------------------------------------------------
__global__ void __launch_bounds__(256)
roipool_kernel(const float* __restrict__ feat,   // [1][256][50][50]
               const float* __restrict__ rois,   // [256][5]
               __bf16* __restrict__ A0)          // [256][12544] bf16
{
    const int bin = blockIdx.x % (ROI_SIZE * ROI_SIZE);
    const int roi = blockIdx.x / (ROI_SIZE * ROI_SIZE);
    const int ph = bin / ROI_SIZE;
    const int pw = bin % ROI_SIZE;
    const int c  = threadIdx.x;

    const float s = 1.0f / 16.0f;
    const float x1 = rintf(rois[roi * 5 + 1] * s);
    const float y1 = rintf(rois[roi * 5 + 2] * s);
    const float x2 = rintf(rois[roi * 5 + 3] * s);
    const float y2 = rintf(rois[roi * 5 + 4] * s);
    const float rw = fmaxf(x2 - x1 + 1.0f, 1.0f);
    const float rh = fmaxf(y2 - y1 + 1.0f, 1.0f);
    const float bh = rh / (float)ROI_SIZE;
    const float bw = rw / (float)ROI_SIZE;

    const int hs = (int)fminf(fmaxf(floorf((float)ph * bh) + y1, 0.0f), (float)HF);
    const int he = (int)fminf(fmaxf(ceilf((float)(ph + 1) * bh) + y1, 0.0f), (float)HF);
    const int ws = (int)fminf(fmaxf(floorf((float)pw * bw) + x1, 0.0f), (float)WF);
    const int we = (int)fminf(fmaxf(ceilf((float)(pw + 1) * bw) + x1, 0.0f), (float)WF);

    const float* fp = feat + (size_t)c * (HF * WF);
    float m = -FLT_MAX;
    for (int h = hs; h < he; ++h)
        for (int w = ws; w < we; ++w)
            m = fmaxf(m, fp[h * WF + w]);
    if (hs >= he || ws >= we) m = 0.0f;   // empty bin -> 0

    A0[(size_t)roi * IN_FLAT + c * (ROI_SIZE * ROI_SIZE) + ph * ROI_SIZE + pw] = (__bf16)m;
}

// ---------------------------------------------------------------------------
// Kernel 2: D = A(bf16)[M][K] * B(f32)[K][N] + bias via WMMA bf16.
//
// WG = 256 threads = 8 waves, tile 128(M) x 64(N); wave = 32x32 (4 accums).
// K-step = 64, LDS double-buffered: stage strip i+1 (64K x 64N, f32->bf16,
// transposed) while computing 8 WMMAs on strip i. One barrier per step.
//
// LDS layout Bt[buf][n][k]: row stride 72 bf16 (144 B) -> fragment reads are
// 16B-aligned ds_load_b128, conflict-free (36-DW row stride).
//
// A-fragment (16-bit A 16x32): lane l = row (l&15); K chunks (l>>4)*8 and
//   16+(l>>4)*8 -> two 16 B global loads.
// B-fragment (16-bit B 32x16): lane l = col (l&15); K chunks (l>>4)*8 and 16+(l>>4)*8
//   in LDS row -> 2 x ds_load_b128.
// C/D f32 16x16: vgpr j: lanes 0-15 (M=j, N=lane); lanes 16-31 (M=8+j).
// ---------------------------------------------------------------------------
#define LDSB_STRIDE 72           // 64 k + 8 pad (bf16 elems), 144 B
#define LDSB_BUF    (64 * LDSB_STRIDE)

template <bool OUT_BF16, bool NALIGN>
__global__ void __launch_bounds__(256)
gemm_bias_kernel(const __bf16* __restrict__ A,    // [M][K] bf16, K % 64 == 0
                 const float*  __restrict__ B,    // [K][N] f32
                 const float*  __restrict__ bias, // [N]
                 float*  __restrict__ outF,       // [M][N] if !OUT_BF16
                 __bf16* __restrict__ outH,       // [M][N] if OUT_BF16
                 int M, int N, int K)
{
    __shared__ __bf16 Bt[2 * LDSB_BUF];

    const int t    = threadIdx.x;
    const int lane = t & 31;
    const int wv   = t >> 5;
    const int wm   = (wv >> 1) * 32;
    const int wn   = (wv & 1)  * 32;
    const int lrow = lane & 15;
    const int lh   = lane >> 4;

    const int mb = blockIdx.y * 128;
    const int nb = blockIdx.x * 64;

    // Staging assignment: thread -> 4 n-cols (bc..bc+3) x 4 k-rows (bk..bk+3)
    const int bc = (t & 15) * 4;
    const int bk = (t >> 4) * 4;

    v8f acc[2][2] = {};

    // ---- stage strip(k0) into buffer ----
    auto stage = [&](int k0, int buf) {
        __bf16* dst = &Bt[buf * LDSB_BUF];
        if (NALIGN) {
            const float* src = B + (size_t)(k0 + bk) * N + nb + bc;
            if (k0 + 128 < K)
                __builtin_prefetch(src + (size_t)128 * N, 0, 1);
            v4f r[4];
            #pragma unroll
            for (int j = 0; j < 4; ++j)
                r[j] = *(const v4f*)(src + (size_t)j * N);
            #pragma unroll
            for (int i = 0; i < 4; ++i) {
                union { __bf16 h[4]; unsigned long long u; } p;
                p.h[0] = (__bf16)r[0][i];
                p.h[1] = (__bf16)r[1][i];
                p.h[2] = (__bf16)r[2][i];
                p.h[3] = (__bf16)r[3][i];
                *(unsigned long long*)&dst[(size_t)(bc + i) * LDSB_STRIDE + bk] = p.u;
            }
        } else {
            #pragma unroll
            for (int j = 0; j < 4; ++j) {
                #pragma unroll
                for (int i = 0; i < 4; ++i) {
                    const int n = nb + bc + i;
                    const float v =
                        (n < N) ? B[(size_t)(k0 + bk + j) * N + n] : 0.0f;
                    dst[(size_t)(bc + i) * LDSB_STRIDE + bk + j] = (__bf16)v;
                }
            }
        }
    };

    stage(0, 0);
    __syncthreads();

    for (int k0 = 0; k0 < K; k0 += 64) {
        const int cur = (k0 >> 6) & 1;
        if (k0 + 64 < K) stage(k0 + 64, cur ^ 1);

        const __bf16* bbase = &Bt[cur * LDSB_BUF];
        #pragma unroll
        for (int kk = 0; kk < 64; kk += 32) {
            // A fragments (direct global, L2-resident)
            v16bf af[2];
            #pragma unroll
            for (int fm = 0; fm < 2; ++fm) {
                const __bf16* ap =
                    A + (size_t)(mb + wm + fm * 16 + lrow) * K + k0 + kk + lh * 8;
                const v8bf a0 = *(const v8bf*)(ap);
                const v8bf a1 = *(const v8bf*)(ap + 16);
                #pragma unroll
                for (int i = 0; i < 8; ++i) { af[fm][i] = a0[i]; af[fm][i + 8] = a1[i]; }
            }
            // B fragments (two 16 B chunks in LDS row: K lh*8 and 16+lh*8)
            v16bf bfr[2];
            #pragma unroll
            for (int fn = 0; fn < 2; ++fn) {
                const __bf16* bp =
                    bbase + (size_t)(wn + fn * 16 + lrow) * LDSB_STRIDE + kk + lh * 8;
                const v8bf b0 = *(const v8bf*)(bp);
                const v8bf b1 = *(const v8bf*)(bp + 16);
                #pragma unroll
                for (int i = 0; i < 8; ++i) { bfr[fn][i] = b0[i]; bfr[fn][i + 8] = b1[i]; }
            }
            #pragma unroll
            for (int fm = 0; fm < 2; ++fm)
                #pragma unroll
                for (int fn = 0; fn < 2; ++fn)
                    acc[fm][fn] = __builtin_amdgcn_wmma_f32_16x16x32_bf16(
                        false, af[fm], false, bfr[fn],
                        (short)0, acc[fm][fn], false, false);
        }
        __syncthreads();
    }

    // ---- epilogue: bias add + store ----
    #pragma unroll
    for (int fn = 0; fn < 2; ++fn) {
        const int n = nb + wn + fn * 16 + lrow;
        if (n >= N) continue;
        const float bv = bias[n];
        #pragma unroll
        for (int fm = 0; fm < 2; ++fm) {
            #pragma unroll
            for (int j = 0; j < 8; ++j) {
                const int m = mb + wm + fm * 16 + lh * 8 + j;
                const float v = acc[fm][fn][j] + bv;
                if (OUT_BF16) outH[(size_t)m * N + n] = (__bf16)v;
                else          outF[(size_t)m * N + n] = v;
            }
        }
    }
}

// ---------------------------------------------------------------------------
// Launcher. Inputs (setup_inputs order):
//  0 features  1 rois  2 W0  3 b0  4 W1  5 b1  6 Wcls  7 bcls  8 Wloc  9 bloc
// Output: cls_locs [256,84] then scores [256,21], flat f32.
// ---------------------------------------------------------------------------
extern "C" void kernel_launch(void* const* d_in, const int* in_sizes, int n_in,
                              void* d_out, int out_size, void* d_ws, size_t ws_size,
                              hipStream_t stream) {
    const float* features = (const float*)d_in[0];
    const float* rois     = (const float*)d_in[1];
    const float* W0   = (const float*)d_in[2];
    const float* b0   = (const float*)d_in[3];
    const float* W1   = (const float*)d_in[4];
    const float* b1   = (const float*)d_in[5];
    const float* Wcls = (const float*)d_in[6];
    const float* bcls = (const float*)d_in[7];
    const float* Wloc = (const float*)d_in[8];
    const float* bloc = (const float*)d_in[9];
    float* out = (float*)d_out;

    // Workspace: bf16 activations only (~8.5 MB)
    char* ws = (char*)d_ws;
    __bf16* A0  = (__bf16*)(ws);                                   // 256x12544
    __bf16* fc1 = (__bf16*)(ws + (size_t)N_ROIS * IN_FLAT * 2);    // 256x2048
    __bf16* fc2 = (__bf16*)(ws + (size_t)N_ROIS * IN_FLAT * 2
                               + (size_t)N_ROIS * HID * 2);        // 256x2048

    // 1) RoIPool -> bf16 activations
    roipool_kernel<<<N_ROIS * ROI_SIZE * ROI_SIZE, 256, 0, stream>>>(
        features, rois, A0);

    // 2) fc1 = pooled @ W0 + b0      (M=256, K=12544, N=2048) -> bf16
    {
        dim3 grid(HID / 64, N_ROIS / 128);
        gemm_bias_kernel<true, true><<<grid, 256, 0, stream>>>(
            A0, W0, b0, nullptr, fc1, N_ROIS, HID, IN_FLAT);
    }
    // 3) fc2 = fc1 @ W1 + b1         (M=256, K=2048, N=2048) -> bf16
    {
        dim3 grid(HID / 64, N_ROIS / 128);
        gemm_bias_kernel<true, true><<<grid, 256, 0, stream>>>(
            fc1, W1, b1, nullptr, fc2, N_ROIS, HID, HID);
    }
    // 4) roi_cls_locs = fc2 @ Wloc + bloc  (N=84) -> f32
    {
        dim3 grid((4 * N_CLASS + 63) / 64, N_ROIS / 128);
        gemm_bias_kernel<false, false><<<grid, 256, 0, stream>>>(
            fc2, Wloc, bloc, out, nullptr, N_ROIS, 4 * N_CLASS, HID);
    }
    // 5) roi_scores = fc2 @ Wcls + bcls    (N=21) -> f32
    {
        dim3 grid((N_CLASS + 63) / 64, N_ROIS / 128);
        gemm_bias_kernel<false, false><<<grid, 256, 0, stream>>>(
            fc2, Wcls, bcls, out + (size_t)N_ROIS * 4 * N_CLASS, nullptr,
            N_ROIS, N_CLASS, HID);
    }
}
